// MoE_13864154432372
// MI455X (gfx1250) — matrix operand order repaired and checked by reading, not compiled, not run on hardware
//
#include <hip/hip_runtime.h>

// ---------------------------------------------------------------------------
// MoE (DeepSeek-V3 style) for MI455X / gfx1250.
//  - bf16 WMMA (v_wmma_f32_16x16x32_bf16), f32 accumulation
//  - weights pre-converted fp32->bf16 once per launch (fits in 192MB L2)
//  - A-tiles staged in LDS, shared by the 8 waves of a workgroup
//  - deterministic: fixed accumulation order, no atomics
// ---------------------------------------------------------------------------

typedef __bf16 bf16;
typedef __attribute__((ext_vector_type(16))) __bf16 v16bf;
typedef __attribute__((ext_vector_type(8)))  __bf16 bf16x8;
typedef __attribute__((ext_vector_type(8)))  float  v8f;

constexpr int   Tn     = 1024;
constexpr int   Dn     = 1024;
constexpr int   En     = 16;
constexpr int   Kn     = 4;
constexpr int   INTERn = 512;
constexpr int   SHIn   = 1024;   // shared expert inter dim
constexpr float RSCALE = 2.5f;

// LDS A-tile: 64 tokens x 32 K, rows padded to 40 elems (80B) for bank spread
constexpr int APAD = 40;

__device__ __forceinline__ float fast_sigmoid(float v) {
  // 1/(1+exp(-v)) via v_rcp_f32 (~1 ulp) -- avoids IEEE div expansion;
  // plenty accurate next to bf16 tensor math.
  return __builtin_amdgcn_rcpf(1.0f + __expf(-v));
}

// ---- WMMA fragment helpers (layouts per CDNA5 ISA §7.12.2) ----

// A fragment from an LDS row: two 16B chunks (K = 2*koff..+7, 2*koff+8..+15)
__device__ __forceinline__ v16bf fragA(const bf16* rowp, int koff) {
  bf16x8 c0 = *(const bf16x8*)(rowp + 2 * koff);
  bf16x8 c1 = *(const bf16x8*)(rowp + 2 * koff + 8);
  v16bf r;
#pragma unroll
  for (int j = 0; j < 8; ++j) { r[j] = c0[j]; r[j + 8] = c1[j]; }
  return r;
}

// B fragment: 16 contiguous K for column n. BT = bf16 (fast path, pure loads)
// or float (fallback, inline convert).
template <typename BT>
__device__ __forceinline__ v16bf loadB(const BT* __restrict__ p) {
  v16bf r;
#pragma unroll
  for (int j = 0; j < 16; ++j) r[j] = (bf16)p[j];
  return r;
}

// ---------------------------------------------------------------------------
// Router: one thread per token. Sigmoid scores, group-limited top-k,
// dense combine[T][E]. First-index tie-break like jax top_k.
// ---------------------------------------------------------------------------
__global__ void __launch_bounds__(256)
route_kernel(const float* __restrict__ x, const float* __restrict__ gw,
             const float* __restrict__ gbias, float* __restrict__ combine) {
  int t = blockIdx.x * blockDim.x + threadIdx.x;
  if (t >= Tn) return;

  const float* xr = x + (size_t)t * Dn;
  float acc[En];
#pragma unroll
  for (int e = 0; e < En; ++e) acc[e] = 0.f;

  for (int d = 0; d < Dn; d += 4) {
    float4 xv = *(const float4*)(xr + d);
#pragma unroll
    for (int e = 0; e < En; ++e) {
      const float* w = gw + (size_t)e * Dn + d;
      acc[e] += xv.x * w[0] + xv.y * w[1] + xv.z * w[2] + xv.w * w[3];
    }
  }

  float sc[En], s[En];
#pragma unroll
  for (int e = 0; e < En; ++e) {
    sc[e] = fast_sigmoid(acc[e]);
    s[e]  = sc[e] + gbias[e];
  }

  float gsc[4];
#pragma unroll
  for (int g = 0; g < 4; ++g) {
    float m1 = -1e30f, m2 = -1e30f;
#pragma unroll
    for (int j = 0; j < 4; ++j) {
      float v = s[g * 4 + j];
      if (v > m1) { m2 = m1; m1 = v; } else if (v > m2) { m2 = v; }
    }
    gsc[g] = m1 + m2;
  }
  int g0 = 0;
  for (int g = 1; g < 4; ++g) if (gsc[g] > gsc[g0]) g0 = g;
  int g1 = -1;
  for (int g = 0; g < 4; ++g)
    if (g != g0 && (g1 < 0 || gsc[g] > gsc[g1])) g1 = g;

  float masked[En];
#pragma unroll
  for (int e = 0; e < En; ++e) {
    int g = e >> 2;
    masked[e] = (g == g0 || g == g1) ? s[e] : -1e30f;
  }

  int   pick[Kn];
  float wsum = 0.f;
#pragma unroll
  for (int k = 0; k < Kn; ++k) {
    int best = 0;
    for (int e = 1; e < En; ++e) if (masked[e] > masked[best]) best = e;
    pick[k] = best;
    masked[best] = -1e30f;
    wsum += sc[best];
  }
  float scale = RSCALE / wsum;   // once per token: keep precise division

  float row[En];
#pragma unroll
  for (int e = 0; e < En; ++e) row[e] = 0.f;
#pragma unroll
  for (int k = 0; k < Kn; ++k) row[pick[k]] = sc[pick[k]] * scale;
#pragma unroll
  for (int e = 0; e < En; ++e) combine[(size_t)t * En + e] = row[e];
}

// ---------------------------------------------------------------------------
// fp32 -> bf16 conversion (float4 per thread)
// ---------------------------------------------------------------------------
__global__ void __launch_bounds__(256)
cvt_kernel(const float* __restrict__ in, bf16* __restrict__ outp, int n4) {
  int i = blockIdx.x * blockDim.x + threadIdx.x;
  if (i >= n4) return;
  float4 v = ((const float4*)in)[i];
  bf16* o = outp + (size_t)i * 4;
  o[0] = (bf16)v.x; o[1] = (bf16)v.y; o[2] = (bf16)v.z; o[3] = (bf16)v.w;
}

// ---------------------------------------------------------------------------
// GEMM1: per-expert  H = combine .* silu(X Gp^T) .* (X Up^T)   (bf16 out)
// Block = (32,8): 8 waves share a 64-token block (A staged in LDS); each
// wave owns one 16-wide inter-column tile. gridDim.z = expert.
// HASC=false -> scale 1.0 (shared-expert instantiation, no branch/loads).
// ---------------------------------------------------------------------------
template <typename BT, bool HASC>
__global__ void __launch_bounds__(256)
mlp_h_kernel(const bf16* __restrict__ xb, const BT* __restrict__ gateW,
             const BT* __restrict__ upW, const float* __restrict__ combine,
             bf16* __restrict__ Hbase, int len) {
  __shared__ bf16 Atile[64][APAD];

  const int lane = threadIdx.x;                  // 0..31
  const int wv   = threadIdx.y;                  // 0..7
  const int tid  = wv * 32 + lane;               // 0..255
  const int e    = blockIdx.z;
  const int i0   = (blockIdx.x * 8 + wv) * 16;   // inter-column tile
  const int t0   = blockIdx.y * 64;              // 64-token block

  const BT* gp = gateW + (size_t)e * len * Dn;
  const BT* up = upW   + (size_t)e * len * Dn;
  bf16*     H  = Hbase + (size_t)e * Tn * len;

  const int mn    = lane & 15;
  const int koff  = (lane >> 4) * 8;    // A fragment K offset
  const int kbase = (lane >> 4) * 16;   // B fragment K offset
  const int roff  = (lane >> 4) * 8;    // C/D row offset

  const int srow = tid >> 2;            // staging: 4 threads per token row
  const int scol = (tid & 3) * 8;       //          8 bf16 (16B) per thread

  v8f accg[4] = {};
  v8f accu[4] = {};

  for (int k0 = 0; k0 < Dn; k0 += 32) {
    __syncthreads();   // previous iteration's fragment reads done
    *(bf16x8*)&Atile[srow][scol] =
        *(const bf16x8*)(xb + (size_t)(t0 + srow) * Dn + k0 + scol);
    __syncthreads();   // tile visible to all waves

    const BT* bgp = gp + (size_t)(i0 + mn) * Dn + k0 + kbase;
    const BT* bup = up + (size_t)(i0 + mn) * Dn + k0 + kbase;
    __builtin_prefetch(bgp + 32, 0, 1);   // speculative: OOB silently dropped
    __builtin_prefetch(bup + 32, 0, 1);
    v16bf bg = loadB(bgp);
    v16bf bu = loadB(bup);
#pragma unroll
    for (int q = 0; q < 4; ++q) {
      v16bf a = fragA(&Atile[q * 16 + mn][0], koff);
      accg[q] = __builtin_amdgcn_wmma_f32_16x16x32_bf16(
          false, a, false, bg, (short)0, accg[q], false, false);
      accu[q] = __builtin_amdgcn_wmma_f32_16x16x32_bf16(
          false, a, false, bu, (short)0, accu[q], false, false);
    }
  }

#pragma unroll
  for (int q = 0; q < 4; ++q) {
#pragma unroll
    for (int r = 0; r < 8; ++r) {
      int   row = t0 + q * 16 + r + roff;
      float cs  = HASC ? combine[(size_t)row * En + e] : 1.0f;
      float g   = accg[q][r];
      float u   = accu[q][r];
      float h   = g * fast_sigmoid(g) * u * cs;   // silu(g)*u*combine
      H[(size_t)row * len + i0 + mn] = (bf16)h;
    }
  }
}

// ---------------------------------------------------------------------------
// GEMM2: out[T,D] = sum over 17 segments (16 routed + shared) of H_seg W_seg^T
// Fixed segment/K order -> deterministic. A (H rows) staged in LDS.
// ---------------------------------------------------------------------------
template <typename BT>
__global__ void __launch_bounds__(256)
down_kernel(const bf16* __restrict__ Hws, const BT* __restrict__ downP,
            const BT* __restrict__ sharedDown, float* __restrict__ out) {
  __shared__ bf16 Atile[64][APAD];

  const int lane = threadIdx.x;
  const int wv   = threadIdx.y;
  const int tid  = wv * 32 + lane;
  const int d0   = (blockIdx.x * 8 + wv) * 16;
  const int t0   = blockIdx.y * 64;

  const int mn    = lane & 15;
  const int koff  = (lane >> 4) * 8;
  const int kbase = (lane >> 4) * 16;
  const int roff  = (lane >> 4) * 8;

  const int srow = tid >> 2;
  const int scol = (tid & 3) * 8;

  v8f acc[4] = {};

  for (int e = 0; e < En + 1; ++e) {
    const int   len = (e < En) ? INTERn : SHIn;
    const bf16* h   = Hws + ((e < En) ? (size_t)e * Tn * INTERn
                                      : (size_t)En * Tn * INTERn);
    const BT*   w   = (e < En) ? (downP + (size_t)e * Dn * INTERn) : sharedDown;

    for (int k0 = 0; k0 < len; k0 += 32) {
      __syncthreads();
      *(bf16x8*)&Atile[srow][scol] =
          *(const bf16x8*)(h + (size_t)(t0 + srow) * len + k0 + scol);
      __syncthreads();

      const BT* bp = w + (size_t)(d0 + mn) * len + k0 + kbase;
      __builtin_prefetch(bp + 32, 0, 1);
      v16bf b = loadB(bp);
#pragma unroll
      for (int q = 0; q < 4; ++q) {
        v16bf a = fragA(&Atile[q * 16 + mn][0], koff);
        acc[q] = __builtin_amdgcn_wmma_f32_16x16x32_bf16(
            false, a, false, b, (short)0, acc[q], false, false);
      }
    }
  }

#pragma unroll
  for (int q = 0; q < 4; ++q)
#pragma unroll
    for (int r = 0; r < 8; ++r)
      out[(size_t)(t0 + q * 16 + r + roff) * Dn + d0 + mn] = acc[q][r];
}

// ---------------------------------------------------------------------------
extern "C" void kernel_launch(void* const* d_in, const int* in_sizes, int n_in,
                              void* d_out, int out_size, void* d_ws, size_t ws_size,
                              hipStream_t stream) {
  const float* x           = (const float*)d_in[0];
  // d_in[1] = token_mask (all true; unused by reference math)
  const float* gate_w      = (const float*)d_in[2];
  const float* gate_bias   = (const float*)d_in[3];
  const float* gate_projs  = (const float*)d_in[4];   // [E, INTER, D]
  const float* up_projs    = (const float*)d_in[5];   // [E, INTER, D]
  const float* down_projs  = (const float*)d_in[6];   // [E, D, INTER]
  const float* shared_gate = (const float*)d_in[7];   // [SHI, D]
  const float* shared_up   = (const float*)d_in[8];   // [SHI, D]
  const float* shared_down = (const float*)d_in[9];   // [D, SHI]
  float*       out         = (float*)d_out;

  // ---- workspace layout -------------------------------------------------
  const size_t routedW = (size_t)En * INTERn * Dn;     // 8.4M elems
  const size_t sharedW = (size_t)SHIn * Dn;            // 1M elems
  char*  ws      = (char*)d_ws;
  size_t off     = 0;
  float* combine = (float*)(ws + off); off += (size_t)Tn * En * 4;     // 64KB
  off = (off + 255) & ~(size_t)255;
  bf16*  xb      = (bf16*)(ws + off);  off += (size_t)Tn * Dn * 2;     // 2MB
  bf16*  Hws     = (bf16*)(ws + off);
  off += ((size_t)En * Tn * INTERn + (size_t)Tn * SHIn) * 2;           // 18MB
  bf16* gateWb  = (bf16*)(ws + off); off += routedW * 2;               // 16.8MB
  bf16* upWb    = (bf16*)(ws + off); off += routedW * 2;
  bf16* downWb  = (bf16*)(ws + off); off += routedW * 2;
  bf16* sGateb  = (bf16*)(ws + off); off += sharedW * 2;               // 2MB
  bf16* sUpb    = (bf16*)(ws + off); off += sharedW * 2;
  bf16* sDownb  = (bf16*)(ws + off); off += sharedW * 2;
  const size_t fullNeed = off;

  dim3 blk(32, 8);

  // 1) routing -> combine[T][E]
  route_kernel<<<dim3(Tn / 256), dim3(256), 0, stream>>>(x, gate_w, gate_bias,
                                                         combine);
  // 2) x -> bf16
  cvt_kernel<<<dim3((Tn * Dn / 4) / 256), dim3(256), 0, stream>>>(x, xb,
                                                                  Tn * Dn / 4);

  if (ws_size >= fullNeed) {
    // ---- fast path: pre-convert all weights to bf16 (L2-resident) -------
    const int rw4 = (int)(routedW / 4), sw4 = (int)(sharedW / 4);
    cvt_kernel<<<dim3(rw4 / 256), dim3(256), 0, stream>>>(gate_projs, gateWb, rw4);
    cvt_kernel<<<dim3(rw4 / 256), dim3(256), 0, stream>>>(up_projs,   upWb,   rw4);
    cvt_kernel<<<dim3(rw4 / 256), dim3(256), 0, stream>>>(down_projs, downWb, rw4);
    cvt_kernel<<<dim3(sw4 / 256), dim3(256), 0, stream>>>(shared_gate, sGateb, sw4);
    cvt_kernel<<<dim3(sw4 / 256), dim3(256), 0, stream>>>(shared_up,   sUpb,   sw4);
    cvt_kernel<<<dim3(sw4 / 256), dim3(256), 0, stream>>>(shared_down, sDownb, sw4);

    mlp_h_kernel<bf16, true><<<dim3(INTERn / 128, Tn / 64, En), blk, 0, stream>>>(
        xb, gateWb, upWb, combine, Hws, INTERn);
    mlp_h_kernel<bf16, false><<<dim3(SHIn / 128, Tn / 64, 1), blk, 0, stream>>>(
        xb, sGateb, sUpb, nullptr, Hws + (size_t)En * Tn * INTERn, SHIn);
    down_kernel<bf16><<<dim3(Dn / 128, Tn / 64), blk, 0, stream>>>(
        Hws, downWb, sDownb, out);
  } else {
    // ---- fallback: convert weights inline in the GEMMs ------------------
    mlp_h_kernel<float, true><<<dim3(INTERn / 128, Tn / 64, En), blk, 0, stream>>>(
        xb, gate_projs, up_projs, combine, Hws, INTERn);
    mlp_h_kernel<float, false><<<dim3(SHIn / 128, Tn / 64, 1), blk, 0, stream>>>(
        xb, shared_gate, shared_up, nullptr, Hws + (size_t)En * Tn * INTERn,
        SHIn);
    down_kernel<float><<<dim3(Dn / 128, Tn / 64), blk, 0, stream>>>(
        Hws, down_projs, shared_down, out);
  }
}
